// THLModel_12369505812859
// MI455X (gfx1250) — compile-verified, matrix operand order checked by reading, not run on hardware
//
#include <hip/hip_runtime.h>
#include <stdint.h>

// ---------------------------------------------------------------------------
// Model constants (match reference)
// ---------------------------------------------------------------------------
#define EDIM   256
#define DQ     128
#define DM     256
#define SLOTS  8192
#define WLOC   64
#define KR     8
#define KW     4
#define ODIM   256
#define BSZ    8
#define TLEN   32
#define DECAY  0.99f
#define INV_SQ_DQ 0.08838834764831845f   // 1/sqrt(128)
#define INV_SQ_DM 0.0625f                // 1/sqrt(256)
#define NPART  8                          // scoring blocks per batch
#define PSLOT  (SLOTS / NPART)            // 1024 slots per scoring block

typedef __bf16 bf16;
typedef __attribute__((ext_vector_type(16))) __bf16 v16bf;
typedef __attribute__((ext_vector_type(8)))  float  v8f;

// ---------------------------------------------------------------------------
// WMMA 16x16x32 bf16 GEMM helper.
//   C[8][N] (fp32, LDS) = A[8][K] (bf16, LDS) x B[K][N], with BT[N][K] (bf16,
//   global, pre-transposed). M is padded 8->16: A rows 8..15 alias rows 0..7
//   (garbage only pollutes D rows 8..15, which are never stored).
// Fragment dword map: dwords {0..3, 8..11} + hi8*4 of the 32-element K-row ->
// compiler vectorizes each fragment into two global_load_b128 (verified in
// round-1 disasm).
// ---------------------------------------------------------------------------
__device__ inline v16bf frag_load_row(const bf16* rowp, int hi8) {
  union { v16bf v; unsigned u[8]; } r;
  const unsigned* p = (const unsigned*)rowp;
#pragma unroll
  for (int i = 0; i < 8; ++i) {
    int kk = (i >> 2) * 16 + hi8 * 8 + (i & 3) * 2;   // even element index
    r.u[i] = p[kk >> 1];
  }
  return r.v;
}

__device__ inline void gemm8(const bf16* A, int K, const bf16* BT, int N, float* C) {
  const int wave   = threadIdx.x >> 5;
  const int nwaves = blockDim.x >> 5;
  const int lane   = threadIdx.x & 31;
  const int r16    = lane & 15;
  const int hi8    = (lane >> 4) & 1;
  const bf16* arow = A + (size_t)(r16 & 7) * K;       // rows 8..15 alias 0..7
  for (int nt = wave * 16; nt < N; nt += nwaves * 16) {
    const bf16* brow = BT + (size_t)(nt + r16) * K;
    v8f acc = {0.f, 0.f, 0.f, 0.f, 0.f, 0.f, 0.f, 0.f};
    for (int k0 = 0; k0 < K; k0 += 32) {
      v16bf a = frag_load_row(arow + k0, hi8);
      v16bf b = frag_load_row(brow + k0, hi8);
      acc = __builtin_amdgcn_wmma_f32_16x16x32_bf16(
          false, a, false, b, (short)0, acc, false, false);
    }
    if (hi8 == 0) {                                    // lanes 0..15 hold M=0..7
#pragma unroll
      for (int r = 0; r < 8; ++r) C[(size_t)r * N + nt + r16] = acc[r];
    }
  }
}

__device__ inline float sigmoidf_(float x) { return 1.f / (1.f + __expf(-x)); }

__device__ inline float dot256(const float4* __restrict__ row,
                               const float4* __restrict__ q) {
  float acc = 0.f;
#pragma unroll 8
  for (int j = 0; j < 64; ++j) {
    float4 m = row[j], qv = q[j];
    acc = fmaf(m.x, qv.x, fmaf(m.y, qv.y, fmaf(m.z, qv.z, fmaf(m.w, qv.w, acc))));
  }
  return acc;
}

// ---------------------------------------------------------------------------
// Prep kernels: state init + fp32 -> bf16 transposed weight staging
// ---------------------------------------------------------------------------
__global__ void copyf(float* dst, const float* src, int n) {
  for (int i = blockIdx.x * blockDim.x + threadIdx.x; i < n; i += gridDim.x * blockDim.x)
    dst[i] = src[i];
}
__global__ void zerof(float* dst, int n) {
  for (int i = blockIdx.x * blockDim.x + threadIdx.x; i < n; i += gridDim.x * blockDim.x)
    dst[i] = 0.f;
}
__global__ void zeroi(int* dst, int n) {
  for (int i = blockIdx.x * blockDim.x + threadIdx.x; i < n; i += gridDim.x * blockDim.x)
    dst[i] = 0;
}
__global__ void cconv(bf16* dst, const float* src, int n) {      // plain convert
  for (int i = blockIdx.x * blockDim.x + threadIdx.x; i < n; i += gridDim.x * blockDim.x)
    dst[i] = (bf16)src[i];
}
// dst[(nOff+n)*ldK + kOff + k] = bf16(src[k*N + n])  (transpose-convert)
__global__ void tconv(bf16* dst, int ldK, int kOff, int nOff,
                      const float* src, int K, int N) {
  int tot = K * N;
  for (int i = blockIdx.x * blockDim.x + threadIdx.x; i < tot; i += gridDim.x * blockDim.x) {
    int k = i / N, n = i % N;
    dst[(size_t)(nOff + n) * ldK + kOff + k] = (bf16)src[i];
  }
}

// ---------------------------------------------------------------------------
// kA: [apply previous step's writer top-4 (deferred merge)] + embedding +
//     local attention (associativity-reduced) + router query. One block.
// ---------------------------------------------------------------------------
__global__ __launch_bounds__(256) void kA_apply_dense(
    const int* ids, const float* emb, float* bufw,
    const float* s0w, const float* s2w,
    float* Ew, float* LRw, float* QMw,
    const bf16* WqT, const bf16* WkH, const bf16* WvT, const bf16* WrT,
    float* memw, int* lww, const float* WVECw,
    const float* CWVw, const int* CWIw, int t, int doWriter) {
  __shared__ bf16  sA[8 * 768];     // WMMA A staging (bf16)
  __shared__ float sO[8 * 256];     // GEMM output / temp
  __shared__ float sE[8 * 256];     // e (fp32)
  __shared__ float sKQ[8 * 256];    // Wk @ q
  __shared__ float sAt[8 * 64];     // attn weights
  __shared__ float gv[8][KW];
  __shared__ int   gi[8][KW];
  const int tid = threadIdx.x;

  // ---- deferred writer apply for step t-1 ----
  if (doWriter) {
    if (tid < 8) {                                    // per-batch merge of 8x4
      float lv[NPART * KW]; int li[NPART * KW];
      for (int c = 0; c < NPART * KW; ++c) {
        lv[c] = CWVw[((size_t)tid * NPART + (c >> 2)) * KW + (c & 3)];
        li[c] = CWIw[((size_t)tid * NPART + (c >> 2)) * KW + (c & 3)];
      }
      for (int k = 0; k < KW; ++k) {
        float bv = -1e30f; int bi = 0x7fffffff, bc = 0;
        for (int c = 0; c < NPART * KW; ++c)
          if (lv[c] > bv || (lv[c] == bv && li[c] < bi)) { bv = lv[c]; bi = li[c]; bc = c; }
        gv[tid][k] = bv; gi[tid][k] = bi; lv[bc] = -3.0e38f;
      }
    }
    __syncthreads();
    for (int i = tid; i < 8 * KW * 256; i += 256) {   // gated scatter write
      int b = i >> 10, k = (i >> 8) & 3, d = i & 255;
      float g = sigmoidf_(gv[b][k]);
      float* row = memw + ((size_t)b * SLOTS + gi[b][k]) * DM;
      row[d] = (1.f - g) * row[d] + g * WVECw[b * 256 + d];
    }
    if (tid < 8 * KW)
      lww[(size_t)(tid >> 2) * SLOTS + gi[tid >> 2][tid & 3]] = t - 1;
  }

  // ---- embedding gather ----
  for (int i = tid; i < 8 * 256; i += 256) {
    int b = i >> 8, d = i & 255;
    float v = emb[(size_t)ids[b * TLEN + t] * EDIM + d];
    sE[i] = v;  Ew[i] = v;
  }
  __syncthreads();
  for (int i = tid; i < 8 * 256; i += 256) sA[i] = (bf16)sE[i];
  __syncthreads();
  gemm8(sA, 256, WqT, 128, sO);                       // q = E @ Wq   [8][128]
  __syncthreads();
  for (int i = tid; i < 8 * 128; i += 256) sA[i] = (bf16)sO[i];
  __syncthreads();
  gemm8(sA, 128, WkH, 256, sKQ);                      // kq = q @ Wk^T [8][256]
  __syncthreads();

  // sc[b][w] = buf[b][w] . kq[b]  (valid mask w < t; T=32 so no wrap)
  for (int i = tid; i < 8 * 64; i += 256) {
    int b = i >> 6, w = i & 63;
    float acc = dot256((const float4*)(bufw + ((size_t)b * WLOC + w) * EDIM),
                       (const float4*)(sKQ + b * 256));
    sAt[i] = (w < t) ? acc * INV_SQ_DQ : -1e9f;
  }
  __syncthreads();
  if (tid < 8) {                                      // softmax over 64
    float mx = -1e30f;
    for (int w = 0; w < 64; ++w) mx = fmaxf(mx, sAt[tid * 64 + w]);
    float ssum = 0.f;
    for (int w = 0; w < 64; ++w) { float e = __expf(sAt[tid * 64 + w] - mx); sAt[tid * 64 + w] = e; ssum += e; }
    float inv = 1.f / ssum;
    for (int w = 0; w < 64; ++w) sAt[tid * 64 + w] *= inv;
  }
  __syncthreads();
  for (int i = tid; i < 8 * 256; i += 256) {          // ab = attn @ buf
    int b = i >> 8, d = i & 255;
    float acc = 0.f;
    for (int w = 0; w < 64; ++w)
      acc = fmaf(sAt[b * 64 + w], bufw[((size_t)b * WLOC + w) * EDIM + d], acc);
    sO[i] = acc;
  }
  __syncthreads();
  for (int i = tid; i < 8 * 256; i += 256) sA[i] = (bf16)sO[i];
  __syncthreads();
  gemm8(sA, 256, WvT, 256, sO);                       // local_read = ab @ Wv
  __syncthreads();
  float lrflag = (t > 0) ? 1.f : 0.f;
  for (int i = tid; i < 8 * 256; i += 256) LRw[i] = sO[i] * lrflag;
  for (int i = tid; i < 8 * 256; i += 256) {          // circular-buffer write
    int b = i >> 8, d = i & 255;
    bufw[((size_t)b * WLOC + (t & (WLOC - 1))) * EDIM + d] = sE[i];
  }
  // U = [e, s0, s2] -> qm = U @ Wr
  for (int i = tid; i < 8 * 768; i += 256) {
    int b = i / 768, c = i % 768;
    float v = (c < 256) ? sE[b * 256 + c]
            : (c < 512) ? s0w[b * 256 + (c - 256)]
                        : s2w[b * 256 + (c - 512)];
    sA[b * 768 + c] = (bf16)v;
  }
  __syncthreads();
  gemm8(sA, 768, WrT, 256, sO);
  __syncthreads();
  for (int i = tid; i < 8 * 256; i += 256) QMw[i] = sO[i];
}

// ---------------------------------------------------------------------------
// kB: router partial scoring. grid = 64 blocks (8 per batch), 1024 slots each.
//     Emits per-block local top-8 candidates; decays rema of owned range.
// ---------------------------------------------------------------------------
__global__ __launch_bounds__(256) void kB_router_score(
    const float* memw, float* remaw, const float* QMw,
    float* CRVw, int* CRIw) {
  __shared__ float sqm[256];                    // first in LDS (async dst)
  __shared__ float ssc[PSLOT];
  __shared__ float rv[256];
  __shared__ int   ri[256];
  const int b = blockIdx.x >> 3, part = blockIdx.x & 7;
  const int tid = threadIdx.x, base = part * PSLOT;
  const float* memb = memw + (size_t)b * SLOTS * DM;
  float* rem = remaw + (size_t)b * SLOTS;
  const float* qmg = QMw + b * 256;

  // stage query via CDNA5 async global->LDS copy (64 lanes x 16B)
  if (tid < 64) {
    unsigned loff = tid * 16u;                  // byte offset of sqm[tid*4]
    asm volatile("global_load_async_to_lds_b128 %0, %1, off"
                 :: "v"(loff),
                    "v"((unsigned long long)(uintptr_t)(qmg + tid * 4))
                 : "memory");
  }
  asm volatile("s_wait_asynccnt 0" ::: "memory");
  __syncthreads();

  for (int sl = tid; sl < PSLOT; sl += 256) {
    int s = base + sl;
    if (sl + 256 < PSLOT) __builtin_prefetch(memb + (size_t)(s + 256) * DM, 0, 1);
    float acc = dot256((const float4*)(memb + (size_t)s * DM), (const float4*)sqm);
    ssc[sl] = acc * INV_SQ_DM - 0.1f * rem[s];
  }
  __syncthreads();
  for (int k = 0; k < KR; ++k) {                // local top-8, low-index ties
    float bv = -1e30f; int bi = 0x7fffffff;
    for (int sl = tid; sl < PSLOT; sl += 256) {
      float v = ssc[sl];
      if (v > bv || (v == bv && sl < bi)) { bv = v; bi = sl; }
    }
    rv[tid] = bv; ri[tid] = bi;
    __syncthreads();
    for (int off = 128; off > 0; off >>= 1) {
      if (tid < off) {
        float v2 = rv[tid + off]; int i2 = ri[tid + off];
        if (v2 > rv[tid] || (v2 == rv[tid] && i2 < ri[tid])) { rv[tid] = v2; ri[tid] = i2; }
      }
      __syncthreads();
    }
    if (tid == 0) {
      CRVw[((size_t)b * NPART + part) * KR + k] = rv[0];
      CRIw[((size_t)b * NPART + part) * KR + k] = base + ri[0];
      ssc[ri[0]] = -3.0e38f;
    }
    __syncthreads();
  }
  for (int sl = tid; sl < PSLOT; sl += 256) rem[base + sl] *= DECAY;  // decay own range
}

// ---------------------------------------------------------------------------
// kC: router merge + read + rema scatter-add, then tier GRUs (stacked-K WMMA),
//     writer vectors, layernorm + head. One block, 256 threads.
// ---------------------------------------------------------------------------
__global__ __launch_bounds__(256) void kC_tier(
    const float* Ew, const float* LRw,
    float* s0w, float* s1w, float* s2w,
    const float* memw, float* remaw,
    const float* CRVw, const int* CRIw,
    const bf16* W01T, const bf16* W1T, const bf16* W2T,
    const bf16* WWT, const bf16* WoutT,
    const float* b0, const float* b1, const float* b2,
    const float* ln_g, const float* ln_b, const float* bout,
    float* WVECw, float* QWw, float* outp, int t, int do1, int do2) {
  __shared__ bf16  sA[8 * 768];
  __shared__ float sP[8 * 512];
  __shared__ float sS0[8 * 256], sS1[8 * 256], sS2[8 * 256];
  __shared__ float sR[8 * 256];
  __shared__ float topv[8][KR], alpha[8][KR];
  __shared__ int   topi[8][KR];
  const int tid = threadIdx.x;

  // ---- merge router candidates (8 parts x 8 each) + softmax ----
  if (tid < 8) {
    float lv[NPART * KR]; int li[NPART * KR];
    for (int c = 0; c < NPART * KR; ++c) {
      lv[c] = CRVw[((size_t)tid * NPART + (c >> 3)) * KR + (c & 7)];
      li[c] = CRIw[((size_t)tid * NPART + (c >> 3)) * KR + (c & 7)];
    }
    for (int k = 0; k < KR; ++k) {
      float bv = -1e30f; int bi = 0x7fffffff, bc = 0;
      for (int c = 0; c < NPART * KR; ++c)
        if (lv[c] > bv || (lv[c] == bv && li[c] < bi)) { bv = lv[c]; bi = li[c]; bc = c; }
      topv[tid][k] = bv; topi[tid][k] = bi; lv[bc] = -3.0e38f;
    }
    float mx = topv[tid][0];
    for (int k = 1; k < KR; ++k) mx = fmaxf(mx, topv[tid][k]);
    float s = 0.f;
    for (int k = 0; k < KR; ++k) { alpha[tid][k] = __expf(topv[tid][k] - mx); s += alpha[tid][k]; }
    float inv = 1.f / s;
    for (int k = 0; k < KR; ++k) alpha[tid][k] *= inv;
  }
  __syncthreads();
  if (tid < 8 * KR)                                  // rema scatter-add (distinct idx)
    remaw[(size_t)(tid >> 3) * SLOTS + topi[tid >> 3][tid & 7]] += alpha[tid >> 3][tid & 7];
  for (int i = tid; i < 8 * 256; i += 256) {          // r = sum alpha_k mem[idx_k]
    int b = i >> 8, d = i & 255;
    float acc = 0.f;
    for (int k = 0; k < KR; ++k)
      acc = fmaf(alpha[b][k], memw[((size_t)b * SLOTS + topi[b][k]) * DM + d], acc);
    sR[i] = acc;
  }
  for (int i = tid; i < 8 * 256; i += 256) { sS0[i] = s0w[i]; sS1[i] = s1w[i]; sS2[i] = s2w[i]; }
  __syncthreads();

  // ---- tier 0: pre = [x0, s0] @ [Wx0; Wh0], x0 = [e, r+lr] ----
  for (int i = tid; i < 8 * 768; i += 256) {
    int b = i / 768, c = i % 768;
    float v = (c < 256) ? Ew[b * 256 + c]
            : (c < 512) ? (sR[b * 256 + (c - 256)] + LRw[b * 256 + (c - 256)])
                        : sS0[b * 256 + (c - 512)];
    sA[i] = (bf16)v;
  }
  __syncthreads();
  gemm8(sA, 768, W01T, 512, sP);
  __syncthreads();
  for (int i = tid; i < 8 * 256; i += 256) {
    int b = i >> 8, d = i & 255;
    float z = sigmoidf_(sP[b * 512 + d] + b0[d]);
    float h = tanhf(sP[b * 512 + 256 + d] + b0[256 + d]);
    sS0[i] = (1.f - z) * sS0[i] + z * h;
  }
  __syncthreads();

  if (do1) {                                          // tier 1 (period 2)
    for (int i = tid; i < 8 * 512; i += 256) {
      int b = i / 512, c = i % 512;
      sA[i] = (bf16)((c < 256) ? sS0[b * 256 + c] : sS1[b * 256 + (c - 256)]);
    }
    __syncthreads();
    gemm8(sA, 512, W1T, 512, sP);
    __syncthreads();
    for (int i = tid; i < 8 * 256; i += 256) {
      int b = i >> 8, d = i & 255;
      float z = sigmoidf_(sP[b * 512 + d] + b1[d]);
      float h = tanhf(sP[b * 512 + 256 + d] + b1[256 + d]);
      sS1[i] = (1.f - z) * sS1[i] + z * h;
    }
    __syncthreads();
  }
  if (do2) {                                          // tier 2 (period 4)
    for (int i = tid; i < 8 * 512; i += 256) {
      int b = i / 512, c = i % 512;
      sA[i] = (bf16)((c < 256) ? sS1[b * 256 + c] : sS2[b * 256 + (c - 256)]);
    }
    __syncthreads();
    gemm8(sA, 512, W2T, 512, sP);
    __syncthreads();
    for (int i = tid; i < 8 * 256; i += 256) {
      int b = i >> 8, d = i & 255;
      float z = sigmoidf_(sP[b * 512 + d] + b2[d]);
      float h = tanhf(sP[b * 512 + 256 + d] + b2[256 + d]);
      sS2[i] = (1.f - z) * sS2[i] + z * h;
    }
    __syncthreads();
  }
  for (int i = tid; i < 8 * 256; i += 256) { s0w[i] = sS0[i]; s1w[i] = sS1[i]; s2w[i] = sS2[i]; }

  // ---- writer vectors: [s0n, s2n, r] @ [Ww | Wwq] ----
  for (int i = tid; i < 8 * 768; i += 256) {
    int b = i / 768, c = i % 768;
    float v = (c < 256) ? sS0[b * 256 + c]
            : (c < 512) ? sS2[b * 256 + (c - 256)]
                        : sR[b * 256 + (c - 512)];
    sA[i] = (bf16)v;
  }
  __syncthreads();
  gemm8(sA, 768, WWT, 512, sP);
  __syncthreads();
  for (int i = tid; i < 8 * 256; i += 256) {
    int b = i >> 8, d = i & 255;
    WVECw[i] = tanhf(sP[b * 512 + d]);
    QWw[i]   = sP[b * 512 + 256 + d];
  }
  __syncthreads();

  // ---- layernorm over cat=[s0n,s1n,s2n]: one wave per batch ----
  {
    int b = tid >> 5, ln = tid & 31;
    float sum = 0.f, sq = 0.f;
    for (int c = ln; c < 768; c += 32) {
      float x = (c < 256) ? sS0[b * 256 + c]
              : (c < 512) ? sS1[b * 256 + (c - 256)]
                          : sS2[b * 256 + (c - 512)];
      sum += x; sq += x * x;
    }
    for (int m = 16; m > 0; m >>= 1) { sum += __shfl_xor(sum, m, 32); sq += __shfl_xor(sq, m, 32); }
    float mu = sum * (1.f / 768.f);
    float var = sq * (1.f / 768.f) - mu * mu;
    float rstd = rsqrtf(var + 1e-5f);
    for (int c = ln; c < 768; c += 32) {
      float x = (c < 256) ? sS0[b * 256 + c]
              : (c < 512) ? sS1[b * 256 + (c - 256)]
                          : sS2[b * 256 + (c - 512)];
      sA[b * 768 + c] = (bf16)((x - mu) * rstd * ln_g[c] + ln_b[c]);
    }
  }
  __syncthreads();
  gemm8(sA, 768, WoutT, 256, sP);                     // logits = nrm @ Wout
  __syncthreads();
  for (int i = tid; i < 8 * 256; i += 256) {
    int b = i >> 8, d = i & 255;
    outp[((size_t)b * TLEN + t) * ODIM + d] = sP[b * 256 + d] + bout[d];
  }
}

// ---------------------------------------------------------------------------
// kD: writer partial scoring. grid = 64 blocks; local top-4 candidates only
//     (the gated write is applied by kA of step t+1; the t=31 write is dead).
// ---------------------------------------------------------------------------
__global__ __launch_bounds__(256) void kD_writer_score(
    const float* memw, const float* remaw, const int* lww,
    const float* QWw, float* CWVw, int* CWIw, int t) {
  __shared__ float sq[256];                     // first in LDS (async dst)
  __shared__ float ssc[PSLOT];
  __shared__ float rv[256];
  __shared__ int   ri[256];
  const int b = blockIdx.x >> 3, part = blockIdx.x & 7;
  const int tid = threadIdx.x, base = part * PSLOT;
  const float* memb = memw + (size_t)b * SLOTS * DM;
  const float* rem = remaw + (size_t)b * SLOTS;
  const int* lw = lww + (size_t)b * SLOTS;
  const float* qg = QWw + b * 256;

  if (tid < 64) {
    unsigned loff = tid * 16u;
    asm volatile("global_load_async_to_lds_b128 %0, %1, off"
                 :: "v"(loff),
                    "v"((unsigned long long)(uintptr_t)(qg + tid * 4))
                 : "memory");
  }
  asm volatile("s_wait_asynccnt 0" ::: "memory");
  __syncthreads();

  const float tf = (float)t, tinv = 1.f / (tf + 1.f);
  for (int sl = tid; sl < PSLOT; sl += 256) {
    int s = base + sl;
    if (sl + 256 < PSLOT) __builtin_prefetch(memb + (size_t)(s + 256) * DM, 0, 1);
    float acc = dot256((const float4*)(memb + (size_t)s * DM), (const float4*)sq);
    float stale = (tf - (float)lw[s]) * tinv;
    float under = 1.f / (1.f + rem[s]);
    ssc[sl] = acc * INV_SQ_DM + 0.5f * stale + 0.5f * under;
  }
  __syncthreads();
  for (int k = 0; k < KW; ++k) {
    float bv = -1e30f; int bi = 0x7fffffff;
    for (int sl = tid; sl < PSLOT; sl += 256) {
      float v = ssc[sl];
      if (v > bv || (v == bv && sl < bi)) { bv = v; bi = sl; }
    }
    rv[tid] = bv; ri[tid] = bi;
    __syncthreads();
    for (int off = 128; off > 0; off >>= 1) {
      if (tid < off) {
        float v2 = rv[tid + off]; int i2 = ri[tid + off];
        if (v2 > rv[tid] || (v2 == rv[tid] && i2 < ri[tid])) { rv[tid] = v2; ri[tid] = i2; }
      }
      __syncthreads();
    }
    if (tid == 0) {
      CWVw[((size_t)b * NPART + part) * KW + k] = rv[0];
      CWIw[((size_t)b * NPART + part) * KW + k] = base + ri[0];
      ssc[ri[0]] = -3.0e38f;
    }
    __syncthreads();
  }
}

// ---------------------------------------------------------------------------
// Host: workspace layout + full step sequence (graph-capture safe).
// ---------------------------------------------------------------------------
extern "C" void kernel_launch(void* const* d_in, const int* in_sizes, int n_in,
                              void* d_out, int out_size, void* d_ws, size_t ws_size,
                              hipStream_t stream) {
  const int*   ids   = (const int*)  d_in[0];
  const float* emb   = (const float*)d_in[1];
  const float* Wq    = (const float*)d_in[2];
  const float* Wk    = (const float*)d_in[3];
  const float* Wv    = (const float*)d_in[4];
  const float* Wr    = (const float*)d_in[5];
  const float* Wx0   = (const float*)d_in[6];
  const float* Wh0   = (const float*)d_in[7];
  const float* b0    = (const float*)d_in[8];
  const float* Wx1   = (const float*)d_in[9];
  const float* Wh1   = (const float*)d_in[10];
  const float* b1    = (const float*)d_in[11];
  const float* Wx2   = (const float*)d_in[12];
  const float* Wh2   = (const float*)d_in[13];
  const float* b2    = (const float*)d_in[14];
  const float* Ww    = (const float*)d_in[15];
  const float* Wwq   = (const float*)d_in[16];
  const float* ln_g  = (const float*)d_in[17];
  const float* ln_b  = (const float*)d_in[18];
  const float* Wout  = (const float*)d_in[19];
  const float* boutp = (const float*)d_in[20];
  const float* mem0  = (const float*)d_in[21];
  float* out = (float*)d_out;

  char* ws = (char*)d_ws;
  size_t off = 0;
  auto alloc = [&](size_t bytes) -> void* {
    void* p = ws + off;
    off = (off + bytes + 255) & ~(size_t)255;
    return p;
  };
  float* memW  = (float*)alloc((size_t)BSZ * SLOTS * DM * 4);   // 64 MB
  float* remaW = (float*)alloc((size_t)BSZ * SLOTS * 4);
  int*   lwW   = (int*)  alloc((size_t)BSZ * SLOTS * 4);
  float* bufW  = (float*)alloc((size_t)BSZ * WLOC * EDIM * 4);
  float* s0W   = (float*)alloc(BSZ * EDIM * 4);
  float* s1W   = (float*)alloc(BSZ * EDIM * 4);
  float* s2W   = (float*)alloc(BSZ * EDIM * 4);
  float* EW    = (float*)alloc(BSZ * EDIM * 4);
  float* LRW   = (float*)alloc(BSZ * EDIM * 4);
  float* QMW   = (float*)alloc(BSZ * DM * 4);
  float* WVECW = (float*)alloc(BSZ * DM * 4);
  float* QWW   = (float*)alloc(BSZ * DM * 4);
  float* CRVW  = (float*)alloc((size_t)BSZ * NPART * KR * 4);
  int*   CRIW  = (int*)  alloc((size_t)BSZ * NPART * KR * 4);
  float* CWVW  = (float*)alloc((size_t)BSZ * NPART * KW * 4);
  int*   CWIW  = (int*)  alloc((size_t)BSZ * NPART * KW * 4);
  bf16* WqT   = (bf16*)alloc((size_t)128 * 256 * 2);
  bf16* WkH   = (bf16*)alloc((size_t)256 * 128 * 2);
  bf16* WvT   = (bf16*)alloc((size_t)256 * 256 * 2);
  bf16* WrT   = (bf16*)alloc((size_t)256 * 768 * 2);
  bf16* W01T  = (bf16*)alloc((size_t)512 * 768 * 2);
  bf16* W1T   = (bf16*)alloc((size_t)512 * 512 * 2);
  bf16* W2T   = (bf16*)alloc((size_t)512 * 512 * 2);
  bf16* WWT   = (bf16*)alloc((size_t)512 * 768 * 2);
  bf16* WoutT = (bf16*)alloc((size_t)256 * 768 * 2);
  (void)ws_size; (void)in_sizes; (void)n_in; (void)out_size;   // ~69 MB used

  // -- state init (runs every call: deterministic, survives ws poisoning) --
  copyf<<<4096, 256, 0, stream>>>(memW, mem0, BSZ * SLOTS * DM);
  zerof<<<64,   256, 0, stream>>>(remaW, BSZ * SLOTS);
  zeroi<<<64,   256, 0, stream>>>(lwW,   BSZ * SLOTS);
  zerof<<<128,  256, 0, stream>>>(bufW,  BSZ * WLOC * EDIM);
  zerof<<<8,    256, 0, stream>>>(s0W, BSZ * EDIM);
  zerof<<<8,    256, 0, stream>>>(s1W, BSZ * EDIM);
  zerof<<<8,    256, 0, stream>>>(s2W, BSZ * EDIM);
  // -- bf16 weight staging (transposed for contiguous WMMA B-fragments) --
  cconv<<<128, 256, 0, stream>>>(WkH, Wk, 256 * 128);
  tconv<<<128, 256, 0, stream>>>(WqT,  256, 0,   0,   Wq,  256, 128);
  tconv<<<256, 256, 0, stream>>>(WvT,  256, 0,   0,   Wv,  256, 256);
  tconv<<<768, 256, 0, stream>>>(WrT,  768, 0,   0,   Wr,  768, 256);
  tconv<<<1024,256, 0, stream>>>(W01T, 768, 0,   0,   Wx0, 512, 512);
  tconv<<<512, 256, 0, stream>>>(W01T, 768, 512, 0,   Wh0, 256, 512);
  tconv<<<512, 256, 0, stream>>>(W1T,  512, 0,   0,   Wx1, 256, 512);
  tconv<<<512, 256, 0, stream>>>(W1T,  512, 256, 0,   Wh1, 256, 512);
  tconv<<<512, 256, 0, stream>>>(W2T,  512, 0,   0,   Wx2, 256, 512);
  tconv<<<512, 256, 0, stream>>>(W2T,  512, 256, 0,   Wh2, 256, 512);
  tconv<<<768, 256, 0, stream>>>(WWT,  768, 0,   0,   Ww,  768, 256);
  tconv<<<768, 256, 0, stream>>>(WWT,  768, 0,   256, Wwq, 768, 256);
  tconv<<<768, 256, 0, stream>>>(WoutT,768, 0,   0,   Wout,768, 256);

  // -- recurrence: stream order provides step sequencing --
  for (int t = 0; t < TLEN; ++t) {
    kA_apply_dense<<<1, 256, 0, stream>>>(ids, emb, bufW, s0W, s2W, EW, LRW, QMW,
                                          WqT, WkH, WvT, WrT,
                                          memW, lwW, WVECW, CWVW, CWIW,
                                          t, (t > 0) ? 1 : 0);
    kB_router_score<<<BSZ * NPART, 256, 0, stream>>>(memW, remaW, QMW, CRVW, CRIW);
    kC_tier<<<1, 256, 0, stream>>>(EW, LRW, s0W, s1W, s2W, memW, remaW, CRVW, CRIW,
                                   W01T, W1T, W2T, WWT, WoutT,
                                   b0, b1, b2, ln_g, ln_b, boutp,
                                   WVECW, QWW, out, t,
                                   (t % 2 == 0) ? 1 : 0, (t % 4 == 0) ? 1 : 0);
    if (t + 1 < TLEN)                           // final writer update is dead code
      kD_writer_score<<<BSZ * NPART, 256, 0, stream>>>(memW, remaW, lwW, QWW,
                                                       CWVW, CWIW, t);
  }
}
